// MultiHeadAttention_77369540870429
// MI455X (gfx1250) — matrix operand order repaired and checked by reading, not compiled, not run on hardware
//
#include <hip/hip_runtime.h>

#define HIDDEN   1024
#define HEADS    16
#define HEAD_DIM 64
#define BATCH    4
#define SEQ      2048

typedef __attribute__((ext_vector_type(16))) __bf16 v16bf;
typedef __attribute__((ext_vector_type(8)))  float  v8f;

#if __has_builtin(__builtin_amdgcn_global_load_async_to_lds_b128)
#define USE_ASYNC 1
#else
#define USE_ASYNC 0
#endif
#if __has_builtin(__builtin_amdgcn_s_wait_asynccnt)
#define HAS_WAIT_ASYNC 1
#else
#define HAS_WAIT_ASYNC 0
#endif
#if __has_builtin(__builtin_amdgcn_tensor_load_to_lds) && \
    __has_builtin(__builtin_amdgcn_s_wait_tensorcnt)
#define USE_TDM 1
#else
#define USE_TDM 0
#endif

// Async builtin prototype (probe-verified round 2): AS1/AS3 pointers to
// gcc-style int4.
typedef int v4i_g __attribute__((vector_size(16)));
typedef __attribute__((address_space(1))) v4i_g as1_v4i;
typedef __attribute__((address_space(3))) v4i_g as3_v4i;
typedef __attribute__((address_space(3))) char as3_char;

// 16-byte global -> LDS copy. Async DMA path (ASYNCcnt) when available.
__device__ __forceinline__ void copy16(const __bf16* gsrc, __bf16* ldst) {
#if USE_ASYNC
  __builtin_amdgcn_global_load_async_to_lds_b128(
      (as1_v4i*)gsrc, (as3_v4i*)ldst, /*imm offset*/0, /*cpol*/0);
#else
  *(uint4*)ldst = *(const uint4*)gsrc;
#endif
}

template <int N>
__device__ __forceinline__ void wait_async() {
#if USE_ASYNC && HAS_WAIT_ASYNC
  __builtin_amdgcn_s_wait_asynccnt(N);
#elif USE_ASYNC
  asm volatile("s_wait_asynccnt %0" ::"i"(N) : "memory");
#endif
}

#if USE_TDM
typedef unsigned tdm_u32x4 __attribute__((vector_size(16)));
typedef int      tdm_i32x8 __attribute__((vector_size(32)));
typedef int      tdm_i32x4 __attribute__((vector_size(16)));

// TDM 2D tile load: 64x64 bf16 tile from a [dim1][dim0] tensor (row stride
// stride0 elements) into LDS with 16B row padding (128B data + 16B pad =>
// 144B = 72 bf16 LDS stride). D# layout per CDNA5 ISA section 8.
// Toolchain builtin is the 6-arg form (round-4 diagnostic):
//   (u32x4 g0, i32x8 g1, i32x4 g2, i32x4 g3, i32x8, i32 cpol)
__device__ __forceinline__ void tdm_load_tile64(const __bf16* gsrc, __bf16* ldst,
                                                unsigned dim0, unsigned dim1,
                                                unsigned stride0) {
  const unsigned lds_addr = (unsigned)(size_t)(as3_char*)ldst;
  const unsigned long long ga = (unsigned long long)(size_t)gsrc;
  const tdm_u32x4 g0 = {
      1u,                                // count=1 (valid), load, not restore
      lds_addr,                          // bits[63:32]  lds_addr
      (unsigned)ga,                      // bits[95:64]  global_addr lo
      (unsigned)(ga >> 32) | (2u << 30)  // bits[120:96] addr hi | type=2
  };
  const tdm_i32x8 g1 = {
      // wg_mask=0 | data_size=1(2B) | pad_enable | pad_interval=4(32DW)
      // | pad_amount=3(4DW)
      (int)((1u << 16) | (1u << 20) | (4u << 22) | (3u << 25)),
      (int)((dim0 & 0xFFFFu) << 16),                    // tensor_dim0 lo16
      (int)((dim0 >> 16) | ((dim1 & 0xFFFFu) << 16)),   // dim0 hi | dim1 lo
      (int)((dim1 >> 16) | (64u << 16)),                // dim1 hi | tile_dim0
      (int)64,                                          // tile_dim1=64, dim2=0
      (int)stride0,                                     // tensor_dim0_stride lo
      0, 0                                              // stride hi, dim1_stride
  };
  const tdm_i32x4 z4 = {0, 0, 0, 0};
  const tdm_i32x8 z8 = {0, 0, 0, 0, 0, 0, 0, 0};
  __builtin_amdgcn_tensor_load_to_lds(g0, g1, z4, z4, z8, 0);
}
#endif

__device__ __forceinline__ v8f wmma_bf16(v16bf a, v16bf b, v8f c) {
  // D = A(16x32 bf16) * B(32x16 bf16) + C(16x16 f32)
  return __builtin_amdgcn_wmma_f32_16x16x32_bf16(
      /*neg_a=*/false, a, /*neg_b=*/false, b,
      /*c_mod=*/(short)0, c, /*reuse_a=*/false, /*reuse_b=*/false);
}

// A fragment: 16x32 bf16, row-major source with leading dim lda (ISA 7.12.2).
__device__ __forceinline__ v16bf load_a_frag(const __bf16* src, int lda) {
  const int lane = threadIdx.x & 31;
  const int half = lane >> 4;
  const int m    = lane & 15;
  const __bf16* row = src + m * lda + half * 8;
  v16bf a;
#pragma unroll
  for (int v = 0; v < 8; ++v) {
    const int k = ((v & 4) << 2) + (v & 3) * 2;
    a[2 * v]     = row[k];
    a[2 * v + 1] = row[k + 1];
  }
  return a;
}

// B fragment: 32x16 bf16 from row-major [K x N] source. Lane%16 = N column, k in table order.
__device__ __forceinline__ v16bf load_b_frag(const __bf16* src, int ldb) {
  const int lane = threadIdx.x & 31;
  const __bf16* colp = src + (lane & 15) + (lane >> 4) * 8 * ldb;
  v16bf b;
#pragma unroll
  for (int i = 0; i < 16; ++i) b[i] = colp[((i & 7) + ((i & 8) << 1)) * ldb];
  return b;
}

// ---------------------------------------------------------------------------
// Kernel 1: QKV projection. C[8192,3072] = query[8192,1024] x w_qkv + b_qkv.
// Epilogue split (Q/K/V and head index) is block-uniform -> scalar branch.
// ---------------------------------------------------------------------------
__global__ __launch_bounds__(256) void qkv_proj_kernel(
    const float* __restrict__ Q, const float* __restrict__ Wqkv,
    const float* __restrict__ Bqkv,
    __bf16* __restrict__ Qh, __bf16* __restrict__ Kt, __bf16* __restrict__ Vh) {
  __shared__ __align__(16) __bf16 As[128 * 34];
  __shared__ __align__(16) __bf16 Bs[32 * 66];
  const int tid  = threadIdx.x;
  const int wave = tid >> 5;
  const int lane = tid & 31;
  const int wm = wave & 3, wn = wave >> 2;  // 4 x 2 wave grid
  const int m0 = blockIdx.y * 128;
  const int n0 = blockIdx.x * 64;

  v8f acc[2][2] = {};

  for (int k0 = 0; k0 < HIDDEN; k0 += 32) {
    // Batch-issue all global loads for this K-step (overlap with barrier wait)
    float4 fa[4], fb[2];
    int ra[4], ca[4], rb[2], cb[2];
#pragma unroll
    for (int i = 0; i < 4; ++i) {
      const int linear = (i * 256 + tid) * 4;
      ra[i] = linear >> 5; ca[i] = linear & 31;
      fa[i] = *(const float4*)(Q + (size_t)(m0 + ra[i]) * HIDDEN + k0 + ca[i]);
    }
#pragma unroll
    for (int i = 0; i < 2; ++i) {
      const int linear = (i * 256 + tid) * 4;
      rb[i] = linear >> 6; cb[i] = linear & 63;
      fb[i] = *(const float4*)(Wqkv + (size_t)(k0 + rb[i]) * (3 * HIDDEN) + n0 + cb[i]);
    }
    __syncthreads();  // previous iteration's fragment reads complete (WAR)
#pragma unroll
    for (int i = 0; i < 4; ++i) {
      __bf16* d = &As[ra[i] * 34 + ca[i]];
      d[0] = (__bf16)fa[i].x; d[1] = (__bf16)fa[i].y;
      d[2] = (__bf16)fa[i].z; d[3] = (__bf16)fa[i].w;
    }
#pragma unroll
    for (int i = 0; i < 2; ++i) {
      __bf16* d = &Bs[rb[i] * 66 + cb[i]];
      d[0] = (__bf16)fb[i].x; d[1] = (__bf16)fb[i].y;
      d[2] = (__bf16)fb[i].z; d[3] = (__bf16)fb[i].w;
    }
    __syncthreads();

    const v16bf a0 = load_a_frag(&As[(wm * 32 + 0) * 34], 34);
    const v16bf a1 = load_a_frag(&As[(wm * 32 + 16) * 34], 34);
    const v16bf b0 = load_b_frag(&Bs[wn * 32 + 0], 66);
    const v16bf b1 = load_b_frag(&Bs[wn * 32 + 16], 66);
    acc[0][0] = wmma_bf16(a0, b0, acc[0][0]);
    acc[0][1] = wmma_bf16(a0, b1, acc[0][1]);
    acc[1][0] = wmma_bf16(a1, b0, acc[1][0]);
    acc[1][1] = wmma_bf16(a1, b1, acc[1][1]);
  }

  const int half = lane >> 4, col = lane & 15;
  const int which = n0 >> 10;          // 0=Q, 1=K, 2=V  (block-uniform)
  const int hh    = (n0 & 1023) >> 6;  // head index     (block-uniform)
#pragma unroll
  for (int t = 0; t < 4; ++t) {
    const int im = t >> 1, in = t & 1;
#pragma unroll
    for (int v = 0; v < 8; ++v) {
      const int gm = m0 + wm * 32 + im * 16 + v + 8 * half;
      const int d  = wn * 32 + in * 16 + col;  // 0..63 within head
      const float val = acc[im][in][v] + Bqkv[n0 + d];
      const int bb = gm >> 11, s = gm & 2047;
      if (which == 0) {
        Qh[((size_t)(bb * HEADS + hh) * SEQ + s) * HEAD_DIM + d] =
            (__bf16)(val * 0.125f);  // 64^-0.5
      } else if (which == 1) {
        Kt[((size_t)(bb * HEADS + hh) * HEAD_DIM + d) * SEQ + s] = (__bf16)val;
      } else {
        Vh[((size_t)(bb * HEADS + hh) * SEQ + s) * HEAD_DIM + d] = (__bf16)val;
      }
    }
  }
}

// ---------------------------------------------------------------------------
// Kernel 2: flash attention per (b, h, 128-row q-tile). 8 waves x 16 q-rows.
// Double-buffered 64x64 K/V tiles staged by the Tensor Data Mover (one D# per
// tile, issued by wave 0, TENSORcnt-tracked), overlapped with WMMA + softmax.
// Fallback: per-lane async global->LDS (ASYNCcnt).
// ---------------------------------------------------------------------------
#define KVSTR 72  // 144B rows: 128B tile row + 16B TDM pad
__global__ __launch_bounds__(256) void attn_kernel(
    const __bf16* __restrict__ Qh, const __bf16* __restrict__ Kt,
    const __bf16* __restrict__ Vh, const float* __restrict__ bias,
    __bf16* __restrict__ ctx) {
  __shared__ __align__(16) __bf16 Ks[2][64 * KVSTR];  // K^T tile: [d][key]
  __shared__ __align__(16) __bf16 Vs[2][64 * KVSTR];  // V tile:   [key][d]
  __shared__ __align__(16) __bf16 Ps[8 * 16 * 66];    // per-wave P: [16 q][64 key]
  const int tid  = threadIdx.x;
  const int wave = tid >> 5, lane = tid & 31;
  const int half = lane >> 4, col = lane & 15;
  const int b = blockIdx.z, h = blockIdx.y;
  const int bh = b * HEADS + h;
  const int q0 = blockIdx.x * 128 + wave * 16;

  const __bf16* Kg = Kt + (size_t)bh * HEAD_DIM * SEQ;
  const __bf16* Vg = Vh + (size_t)bh * SEQ * HEAD_DIM;

#if USE_TDM
  // Scalar wave id -> guaranteed scalar branch (TDM ignores EXEC, so the
  // issue must be skipped by *branch*, not by exec-masking).
  const int wave_s = (int)(__builtin_amdgcn_readfirstlane(tid) >> 5);
  if (wave_s == 0) {
    tdm_load_tile64(Kg, &Ks[0][0], SEQ, HEAD_DIM, SEQ);
    tdm_load_tile64(Vg, &Vs[0][0], HEAD_DIM, SEQ, HEAD_DIM);
  }
#else
  // Per-thread staging coordinates: 2x16B chunks of K and of V per tile
  const int lin0 = tid * 8, lin1 = (256 + tid) * 8;
  const int r0 = lin0 >> 6, c0 = lin0 & 63;
  const int r1 = lin1 >> 6, c1 = lin1 & 63;
  copy16(Kg + (size_t)r0 * SEQ + c0, &Ks[0][r0 * KVSTR + c0]);
  copy16(Vg + (size_t)r0 * HEAD_DIM + c0, &Vs[0][r0 * KVSTR + c0]);
  copy16(Kg + (size_t)r1 * SEQ + c1, &Ks[0][r1 * KVSTR + c1]);
  copy16(Vg + (size_t)r1 * HEAD_DIM + c1, &Vs[0][r1 * KVSTR + c1]);
#endif

  // Q fragments (already scaled by d^-0.5 in kernel 1)
  const __bf16* qbase = Qh + ((size_t)bh * SEQ + q0) * HEAD_DIM;
  const v16bf qa0 = load_a_frag(qbase, HEAD_DIM);       // K-dim 0..31
  const v16bf qa1 = load_a_frag(qbase + 32, HEAD_DIM);  // K-dim 32..63

  v8f o0 = {}, o1 = {}, o2 = {}, o3 = {};
  float mrow[8], lrow[8];
#pragma unroll
  for (int v = 0; v < 8; ++v) { mrow[v] = -1e30f; lrow[v] = 0.f; }

  __bf16* Pw = &Ps[wave * 16 * 66];

  for (int j = 0; j < SEQ / 64; ++j) {
    const int cur = j & 1;
    const int key0 = j * 64;
    // Prefetch tile j+1 into the other buffer (last read 2 iterations ago,
    // protected by the trailing barrier of iteration j-1), then drain tile j.
#if USE_TDM
    if (wave_s == 0) {
      if (j + 1 < SEQ / 64) {
        const int nk = key0 + 64;
        tdm_load_tile64(Kg + nk, &Ks[cur ^ 1][0], SEQ, HEAD_DIM, SEQ);
        tdm_load_tile64(Vg + (size_t)nk * HEAD_DIM, &Vs[cur ^ 1][0],
                        HEAD_DIM, SEQ, HEAD_DIM);
        __builtin_amdgcn_s_wait_tensorcnt(2);  // tile j done, j+1 in flight
      } else {
        __builtin_amdgcn_s_wait_tensorcnt(0);
      }
    }
#else
    if (j + 1 < SEQ / 64) {
      const int nk = key0 + 64;
      __bf16* kd = Ks[cur ^ 1];
      __bf16* vd = Vs[cur ^ 1];
      copy16(Kg + (size_t)r0 * SEQ + nk + c0, &kd[r0 * KVSTR + c0]);
      copy16(Vg + (size_t)(nk + r0) * HEAD_DIM + c0, &vd[r0 * KVSTR + c0]);
      copy16(Kg + (size_t)r1 * SEQ + nk + c1, &kd[r1 * KVSTR + c1]);
      copy16(Vg + (size_t)(nk + r1) * HEAD_DIM + c1, &vd[r1 * KVSTR + c1]);
      wait_async<4>();
    } else {
      wait_async<0>();
    }
#endif
    __syncthreads();  // tile-j staging visible to all waves

    const __bf16* ks = Ks[cur];
    const __bf16* vs = Vs[cur];

    // logits[16 q][64 key] = Q x K^T
    v8f cl0 = {}, cl1 = {}, cl2 = {}, cl3 = {};
    {
      v16bf bb;
      bb = load_b_frag(&ks[0 * KVSTR + 0], KVSTR);   cl0 = wmma_bf16(qa0, bb, cl0);
      bb = load_b_frag(&ks[32 * KVSTR + 0], KVSTR);  cl0 = wmma_bf16(qa1, bb, cl0);
      bb = load_b_frag(&ks[0 * KVSTR + 16], KVSTR);  cl1 = wmma_bf16(qa0, bb, cl1);
      bb = load_b_frag(&ks[32 * KVSTR + 16], KVSTR); cl1 = wmma_bf16(qa1, bb, cl1);
      bb = load_b_frag(&ks[0 * KVSTR + 32], KVSTR);  cl2 = wmma_bf16(qa0, bb, cl2);
      bb = load_b_frag(&ks[32 * KVSTR + 32], KVSTR); cl2 = wmma_bf16(qa1, bb, cl2);
      bb = load_b_frag(&ks[0 * KVSTR + 48], KVSTR);  cl3 = wmma_bf16(qa0, bb, cl3);
      bb = load_b_frag(&ks[32 * KVSTR + 48], KVSTR); cl3 = wmma_bf16(qa1, bb, cl3);
    }

    // Online softmax. C layout: row = v + 8*half (16 lanes/row), col = lane&15.
#pragma unroll
    for (int v = 0; v < 8; ++v) {
      const int qrow = q0 + v + 8 * half;
      const float* bptr = bias + (size_t)qrow * SEQ + key0 + col;
      const float vals0 = cl0[v] + bptr[0];
      const float vals1 = cl1[v] + bptr[16];
      const float vals2 = cl2[v] + bptr[32];
      const float vals3 = cl3[v] + bptr[48];
      float mx = fmaxf(fmaxf(vals0, vals1), fmaxf(vals2, vals3));
      mx = fmaxf(mx, __shfl_xor(mx, 1, 32));
      mx = fmaxf(mx, __shfl_xor(mx, 2, 32));
      mx = fmaxf(mx, __shfl_xor(mx, 4, 32));
      mx = fmaxf(mx, __shfl_xor(mx, 8, 32));
      const float mnew = fmaxf(mrow[v], mx);
      const float sc = __expf(mrow[v] - mnew);
      mrow[v] = mnew;
      float s = 0.f;
      __bf16* prow = &Pw[(v + 8 * half) * 66 + col];
      const float p0 = __expf(vals0 - mnew); s += p0; prow[0]  = (__bf16)p0;
      const float p1 = __expf(vals1 - mnew); s += p1; prow[16] = (__bf16)p1;
      const float p2 = __expf(vals2 - mnew); s += p2; prow[32] = (__bf16)p2;
      const float p3 = __expf(vals3 - mnew); s += p3; prow[48] = (__bf16)p3;
      s += __shfl_xor(s, 1, 32);
      s += __shfl_xor(s, 2, 32);
      s += __shfl_xor(s, 4, 32);
      s += __shfl_xor(s, 8, 32);
      lrow[v] = lrow[v] * sc + s;
      o0[v] *= sc; o1[v] *= sc; o2[v] *= sc; o3[v] *= sc;
    }

    // O += P x V  (per-wave P region; same-wave DS ordering via DScnt)
    {
      const v16bf pa0 = load_a_frag(&Pw[0], 66);   // keys 0..31
      const v16bf pa1 = load_a_frag(&Pw[32], 66);  // keys 32..63
      v16bf bb;
      bb = load_b_frag(&vs[0 * KVSTR + 0], KVSTR);   o0 = wmma_bf16(pa0, bb, o0);
      bb = load_b_frag(&vs[32 * KVSTR + 0], KVSTR);  o0 = wmma_bf16(pa1, bb, o0);
      bb = load_b_frag(&vs[0 * KVSTR + 16], KVSTR);  o1 = wmma_bf16(pa0, bb, o1);
      bb = load_b_frag(&vs[32 * KVSTR + 16], KVSTR); o1 = wmma_bf16(pa1, bb, o1);
      bb = load_b_frag(&vs[0 * KVSTR + 32], KVSTR);  o2 = wmma_bf16(pa0, bb, o2);
      bb = load_b_frag(&vs[32 * KVSTR + 32], KVSTR); o2 = wmma_bf16(pa1, bb, o2);
      bb = load_b_frag(&vs[0 * KVSTR + 48], KVSTR);  o3 = wmma_bf16(pa0, bb, o3);
      bb = load_b_frag(&vs[32 * KVSTR + 48], KVSTR); o3 = wmma_bf16(pa1, bb, o3);
    }
    __syncthreads();  // all waves done reading buf[cur] before tile j+2 staging
  }

  // Finalize: divide by row sum, write combined-head ctx[b][s][h*64+d] in bf16
#pragma unroll
  for (int v = 0; v < 8; ++v) {
    const float inv = 1.0f / lrow[v];
    const int qrow = q0 + v + 8 * half;
    __bf16* dst = ctx + (size_t)(b * SEQ + qrow) * HIDDEN + h * HEAD_DIM + col;
    dst[0]  = (__bf16)(o0[v] * inv);
    dst[16] = (__bf16)(o1[v] * inv);
    dst[32] = (__bf16)(o2[v] * inv);
    dst[48] = (__bf16)(o3[v] * inv);
  }
}

// ---------------------------------------------------------------------------
// Kernel 3: output projection. out[8192,1024] = ctx(bf16) x w_o + b_o (fp32)
// A tile (already bf16) staged via per-lane async global->LDS DMA.
// ---------------------------------------------------------------------------
#define ASTR 40  // 80B rows: 16B-aligned for async b128
__global__ __launch_bounds__(256) void out_proj_kernel(
    const __bf16* __restrict__ Ctx, const float* __restrict__ Wo,
    const float* __restrict__ Bo, float* __restrict__ Out) {
  __shared__ __align__(16) __bf16 As[128 * ASTR];
  __shared__ __align__(16) __bf16 Bs[32 * 66];
  const int tid  = threadIdx.x;
  const int wave = tid >> 5;
  const int lane = tid & 31;
  const int wm = wave & 3, wn = wave >> 2;
  const int m0 = blockIdx.y * 128;
  const int n0 = blockIdx.x * 64;

  // Per-thread A staging coords: 2x16B chunks
  const int lin0 = tid * 8, lin1 = (256 + tid) * 8;
  const int ar0 = lin0 >> 5, ac0 = lin0 & 31;
  const int ar1 = lin1 >> 5, ac1 = lin1 & 31;

  v8f acc[2][2] = {};

  for (int k0 = 0; k0 < HIDDEN; k0 += 32) {
    // Batch B-tile loads (read-only, safe before barrier)
    float4 fb[2];
    int rb[2], cb[2];
#pragma unroll
    for (int i = 0; i < 2; ++i) {
      const int linear = (i * 256 + tid) * 4;
      rb[i] = linear >> 6; cb[i] = linear & 63;
      fb[i] = *(const float4*)(Wo + (size_t)(k0 + rb[i]) * HIDDEN + n0 + cb[i]);
    }
    __syncthreads();  // previous fragment reads complete (WAR)
    // A tile: async bf16 copy straight into LDS
    copy16(Ctx + (size_t)(m0 + ar0) * HIDDEN + k0 + ac0, &As[ar0 * ASTR + ac0]);
    copy16(Ctx + (size_t)(m0 + ar1) * HIDDEN + k0 + ac1, &As[ar1 * ASTR + ac1]);
    // B tile: fp32 -> bf16 stores
#pragma unroll
    for (int i = 0; i < 2; ++i) {
      __bf16* d = &Bs[rb[i] * 66 + cb[i]];
      d[0] = (__bf16)fb[i].x; d[1] = (__bf16)fb[i].y;
      d[2] = (__bf16)fb[i].z; d[3] = (__bf16)fb[i].w;
    }
    wait_async<0>();
    __syncthreads();

    const v16bf a0 = load_a_frag(&As[(wm * 32 + 0) * ASTR], ASTR);
    const v16bf a1 = load_a_frag(&As[(wm * 32 + 16) * ASTR], ASTR);
    const v16bf b0 = load_b_frag(&Bs[wn * 32 + 0], 66);
    const v16bf b1 = load_b_frag(&Bs[wn * 32 + 16], 66);
    acc[0][0] = wmma_bf16(a0, b0, acc[0][0]);
    acc[0][1] = wmma_bf16(a0, b1, acc[0][1]);
    acc[1][0] = wmma_bf16(a1, b0, acc[1][0]);
    acc[1][1] = wmma_bf16(a1, b1, acc[1][1]);
  }

  const int half = lane >> 4, col = lane & 15;
#pragma unroll
  for (int t = 0; t < 4; ++t) {
    const int im = t >> 1, in = t & 1;
#pragma unroll
    for (int v = 0; v < 8; ++v) {
      const int gm = m0 + wm * 32 + im * 16 + v + 8 * half;
      const int gn = n0 + wn * 32 + in * 16 + col;
      Out[(size_t)gm * HIDDEN + gn] = acc[im][in][v] + Bo[gn];
    }
  }
}

// ---------------------------------------------------------------------------
extern "C" void kernel_launch(void* const* d_in, const int* in_sizes, int n_in,
                              void* d_out, int out_size, void* d_ws, size_t ws_size,
                              hipStream_t stream) {
  (void)in_sizes; (void)n_in; (void)out_size; (void)ws_size;
  const float* query = (const float*)d_in[0];
  const float* bias  = (const float*)d_in[1];
  const float* w_qkv = (const float*)d_in[2];
  const float* b_qkv = (const float*)d_in[3];
  const float* w_o   = (const float*)d_in[4];
  const float* b_o   = (const float*)d_in[5];
  float* out = (float*)d_out;

  char* ws = (char*)d_ws;
  const size_t SZ = (size_t)BATCH * HEADS * SEQ * HEAD_DIM * 2;  // 16 MB each
  __bf16* Qh  = (__bf16*)(ws);
  __bf16* Kt  = (__bf16*)(ws + SZ);
  __bf16* Vh  = (__bf16*)(ws + 2 * SZ);
  __bf16* Ctx = (__bf16*)(ws + 3 * SZ);

  qkv_proj_kernel<<<dim3(3 * HIDDEN / 64, BATCH * SEQ / 128), 256, 0, stream>>>(
      query, w_qkv, b_qkv, Qh, Kt, Vh);
  attn_kernel<<<dim3(SEQ / 128, HEADS, BATCH), 256, 0, stream>>>(
      Qh, Kt, Vh, bias, Ctx);
  out_proj_kernel<<<dim3(HIDDEN / 64, BATCH * SEQ / 128), 256, 0, stream>>>(
      Ctx, w_o, b_o, out);
}
